// JointTransformerBlock_31421980738008
// MI455X (gfx1250) — matrix operand order repaired and checked by reading, not compiled, not run on hardware
//
#include <hip/hip_runtime.h>
#include <cstdint>
#include <cstddef>

// ---------------------------------------------------------------------------
// MMDiT JointTransformerBlock for MI455X (gfx1250, wave32, WMMA bf16)
// ---------------------------------------------------------------------------

typedef __attribute__((ext_vector_type(16))) __bf16 bf16x16;
typedef __attribute__((ext_vector_type(8)))  float  f32x8;

static constexpr int   kDim     = 1536;
static constexpr int   kSixDim  = 9216;
static constexpr int   kTotal   = 3072;   // hidden tokens
static constexpr int   kEnc     = 128;
static constexpr int   kStages  = 3;
static constexpr int   kHeads   = 24;
static constexpr int   kHeadDim = 64;
static constexpr int   kInner   = 6144;
static constexpr float kScale   = 0.125f;   // 64^-0.5
static constexpr float kNegInf  = -3.0e38f;

// per-stage geometry
// HIDDEN_LENGTH = (512,1024,1536); S_i = 128 + L_i = 640,1152,1664 ; sum=3456
static constexpr int kRowsAll = 3456;
static constexpr int kQTilesAll = 216;    // 40 + 72 + 104

// ---- workspace layout (bytes, all offsets 256B aligned) -------------------
static constexpr size_t SZ_NH  = (size_t)kTotal * kDim * 4;        // 18874368
static constexpr size_t SZ_NE  = (size_t)kStages * kEnc * kDim * 4;//  2359296
static constexpr size_t SZ_QS  = (size_t)kRowsAll * kDim * 2;      // 10616832

static constexpr size_t OFF_SILU = 0;                        // 3*1536*4
static constexpr size_t OFF_E6   = 32768;                    // 3*9216*4
static constexpr size_t OFF_EC   = OFF_E6 + 110592;
static constexpr size_t OFF_NH   = 262144;
static constexpr size_t OFF_NE   = OFF_NH + SZ_NH;
static constexpr size_t OFF_Q    = OFF_NE + SZ_NE;
static constexpr size_t OFF_K    = OFF_Q  + SZ_NH;
static constexpr size_t OFF_V    = OFF_K  + SZ_NH;
static constexpr size_t OFF_EQ   = OFF_V  + SZ_NH;
static constexpr size_t OFF_EK   = OFF_EQ + SZ_NE;
static constexpr size_t OFF_EV   = OFF_EK + SZ_NE;
static constexpr size_t OFF_QS   = OFF_EV + SZ_NE;
static constexpr size_t OFF_KS   = OFF_QS + SZ_QS;
static constexpr size_t OFF_VS   = OFF_KS + SZ_QS;
static constexpr size_t OFF_AH   = OFF_VS + SZ_QS;           // attn input (hid rows)
static constexpr size_t OFF_AE   = OFF_AH + SZ_NH;           // attn input (enc rows)
// reuse (dead-buffer recycling):
static constexpr size_t OFF_ATTNH = OFF_NH;    // proj(attn) hidden
static constexpr size_t OFF_ATTNE = OFF_NE;    // proj(attn) enc
static constexpr size_t OFF_H1    = OFF_Q;
static constexpr size_t OFF_MOD2  = OFF_K;
static constexpr size_t OFF_FOUT  = OFF_V;
static constexpr size_t OFF_E1    = OFF_EQ;
static constexpr size_t OFF_MODC2 = OFF_EK;
static constexpr size_t OFF_FCOUT = OFF_EV;
static constexpr size_t OFF_FFG   = OFF_QS;                  // bf16 3072x6144
static constexpr size_t OFF_FFCG  = OFF_QS + (size_t)kTotal * kInner * 2;

__device__ __forceinline__ int stage_of_hidden_row(int r) {
  return (r < 512) ? 0 : ((r < 1536) ? 1 : 2);
}

// ---------------------------------------------------------------------------
// 1) silu(temb)
// ---------------------------------------------------------------------------
__global__ __launch_bounds__(256) void silu_kernel(const float* __restrict__ in,
                                                   float* __restrict__ out, int n) {
  int i = blockIdx.x * 256 + threadIdx.x;
  if (i < n) {
    float v = in[i];
    out[i] = v / (1.0f + __expf(-v));
  }
}

// ---------------------------------------------------------------------------
// 2) AdaLN projection: out[3,9216] = silu_temb[3,1536] @ W[1536,9216] + b
//    (tiny GEMM -> plain VALU)
// ---------------------------------------------------------------------------
__global__ __launch_bounds__(256) void adaln_kernel(const float* __restrict__ silu,
                                                    const float* __restrict__ W,
                                                    const float* __restrict__ b,
                                                    float* __restrict__ out) {
  int idx = blockIdx.x * 256 + threadIdx.x;
  if (idx >= kStages * kSixDim) return;
  int r = idx / kSixDim;
  int n = idx - r * kSixDim;
  const float* s = silu + (size_t)r * kDim;
  float acc = b[n];
  for (int k = 0; k < kDim; ++k)
    acc = fmaf(s[k], W[(size_t)k * kSixDim + n], acc);
  out[idx] = acc;
}

// ---------------------------------------------------------------------------
// 3) LayerNorm + AdaLN modulation: out = ln(x)*(1+scale)+shift  (per stage)
// ---------------------------------------------------------------------------
__global__ __launch_bounds__(256) void ln_mod_kernel(const float* __restrict__ x,
                                                     const float* __restrict__ mod,
                                                     float* __restrict__ out,
                                                     int shift_sel, int scale_sel,
                                                     int is_hidden) {
  const int r = blockIdx.x;
  const int stage = is_hidden ? stage_of_hidden_row(r) : (r >> 7);
  const float* xr = x + (size_t)r * kDim;
  float s = 0.f, q = 0.f;
  for (int c = threadIdx.x; c < kDim; c += 256) {
    float v = xr[c]; s += v; q += v * v;
  }
#pragma unroll
  for (int off = 16; off > 0; off >>= 1) {
    s += __shfl_xor(s, off);
    q += __shfl_xor(q, off);
  }
  __shared__ float sA[8], sQ[8];
  if ((threadIdx.x & 31) == 0) { sA[threadIdx.x >> 5] = s; sQ[threadIdx.x >> 5] = q; }
  __syncthreads();
  float ts = 0.f, tq = 0.f;
#pragma unroll
  for (int i = 0; i < 8; ++i) { ts += sA[i]; tq += sQ[i]; }
  const float mean = ts * (1.0f / kDim);
  const float var  = tq * (1.0f / kDim) - mean * mean;
  const float rstd = rsqrtf(var + 1e-6f);
  const float* sh = mod + (size_t)stage * kSixDim + (size_t)shift_sel * kDim;
  const float* sc = mod + (size_t)stage * kSixDim + (size_t)scale_sel * kDim;
  float* orow = out + (size_t)r * kDim;
  for (int c = threadIdx.x; c < kDim; c += 256)
    orow[c] = (xr[c] - mean) * rstd * (1.0f + sc[c]) + sh[c];
}

// ---------------------------------------------------------------------------
// 4) Generic bf16-WMMA GEMM: C[M,N] = A[M,K] @ W[K,N] + bias (opt GELU)
//    BM=BN=128, BK=32, 8 waves, each wave 2x4 16x16 f32 accumulators.
// ---------------------------------------------------------------------------
template <bool A_BF16, bool OUT_BF16, bool GELU>
__global__ __launch_bounds__(256) void gemm_bf16_wmma(const void* __restrict__ Ap,
                                                      const float* __restrict__ W,
                                                      const float* __restrict__ bias,
                                                      void* __restrict__ Cp,
                                                      int M, int N, int K) {
  constexpr int BM = 128, BN = 128, BK = 32;
  __shared__ __bf16 As[BM][BK + 8];
  __shared__ __bf16 Bs[BK][BN + 8];

  const int tid  = threadIdx.x;
  const int lane = tid & 31;
  const int wid  = tid >> 5;
  const int wm   = wid >> 1;   // 0..3  (32 rows / wave)
  const int wn   = wid & 1;    // 0..1  (64 cols / wave)
  const int bm   = blockIdx.y * BM;
  const int bn   = blockIdx.x * BN;
  const int l15  = lane & 15;
  const int lhi  = lane >> 4;

  f32x8 acc[2][4];
#pragma unroll
  for (int i = 0; i < 2; ++i)
#pragma unroll
    for (int j = 0; j < 4; ++j) acc[i][j] = f32x8{0, 0, 0, 0, 0, 0, 0, 0};

  for (int k0 = 0; k0 < K; k0 += BK) {
    // --- stage A tile (128x32), convert f32 (or copy bf16) -> LDS bf16 ----
    {
      const int row = tid >> 3;          // 0..31
      const int c4  = (tid & 7) * 4;     // 0..28
#pragma unroll
      for (int rr = 0; rr < 4; ++rr) {
        const int ar = row + rr * 32;
        if constexpr (A_BF16) {
          const __bf16* A = (const __bf16*)Ap + (size_t)(bm + ar) * K + k0 + c4;
          As[ar][c4 + 0] = A[0]; As[ar][c4 + 1] = A[1];
          As[ar][c4 + 2] = A[2]; As[ar][c4 + 3] = A[3];
        } else {
          const float4 v = *(const float4*)((const float*)Ap + (size_t)(bm + ar) * K + k0 + c4);
          As[ar][c4 + 0] = (__bf16)v.x; As[ar][c4 + 1] = (__bf16)v.y;
          As[ar][c4 + 2] = (__bf16)v.z; As[ar][c4 + 3] = (__bf16)v.w;
        }
      }
    }
    // --- stage B tile (32x128) --------------------------------------------
    {
      const int row = tid >> 5;          // 0..7
      const int c4  = (tid & 31) * 4;    // 0..124
#pragma unroll
      for (int rr = 0; rr < 4; ++rr) {
        const int br = row + rr * 8;
        const float4 v = *(const float4*)(W + (size_t)(k0 + br) * N + bn + c4);
        Bs[br][c4 + 0] = (__bf16)v.x; Bs[br][c4 + 1] = (__bf16)v.y;
        Bs[br][c4 + 2] = (__bf16)v.z; Bs[br][c4 + 3] = (__bf16)v.w;
      }
    }
    // prefetch next K-step (lowers to global_prefetch_b8 on gfx1250)
    if (k0 + BK < K) {
      __builtin_prefetch(W + (size_t)(k0 + BK + (tid >> 5)) * N + bn + (tid & 31) * 4, 0, 1);
      if constexpr (!A_BF16)
        __builtin_prefetch((const float*)Ap + (size_t)(bm + (tid >> 3)) * K + k0 + BK, 0, 1);
    }
    __syncthreads();

    // --- fragments + WMMA --------------------------------------------------
    bf16x16 afrag[2];
#pragma unroll
    for (int mi = 0; mi < 2; ++mi) {
      const int arow = wm * 32 + mi * 16 + l15;
#pragma unroll
      for (int e = 0; e < 16; ++e) {
        const int kk = ((e < 8) ? 0 : 16) + lhi * 8 + (e & 7);
        afrag[mi][e] = As[arow][kk];
      }
    }
#pragma unroll
    for (int ni = 0; ni < 4; ++ni) {
      bf16x16 bfrag;
      const int bcol = wn * 64 + ni * 16 + l15;
#pragma unroll
      for (int e = 0; e < 16; ++e) bfrag[e] = Bs[((e < 8) ? 0 : 16) + lhi * 8 + (e & 7)][bcol];
#pragma unroll
      for (int mi = 0; mi < 2; ++mi) {
        acc[mi][ni] = __builtin_amdgcn_wmma_f32_16x16x32_bf16(
            false, afrag[mi], false, bfrag, (short)0, acc[mi][ni], false, false);
      }
    }
    __syncthreads();
  }

  // --- epilogue -------------------------------------------------------------
#pragma unroll
  for (int mi = 0; mi < 2; ++mi) {
#pragma unroll
    for (int ni = 0; ni < 4; ++ni) {
#pragma unroll
      for (int v = 0; v < 8; ++v) {
        const int r = bm + wm * 32 + mi * 16 + lhi * 8 + v;
        const int c = bn + wn * 64 + ni * 16 + l15;
        if (r < M && c < N) {
          float val = acc[mi][ni][v];
          if (bias) val += bias[c];
          if constexpr (GELU) {
            const float x = val;
            const float u = 0.7978845608028654f * (x + 0.044715f * x * x * x);
            val = 0.5f * x * (1.0f + tanhf(u));
          }
          if constexpr (OUT_BF16)
            ((__bf16*)Cp)[(size_t)r * N + c] = (__bf16)val;
          else
            ((float*)Cp)[(size_t)r * N + c] = val;
        }
      }
    }
  }
}

// ---------------------------------------------------------------------------
// 5) per-head RMSNorm + RoPE + pack f32 -> bf16 into per-stage [S,24,64]
//    One wave per (row, head); lane t owns dims (2t, 2t+1) -> RoPE pair.
// ---------------------------------------------------------------------------
__global__ __launch_bounds__(32) void pack_qkv_kernel(
    const float* __restrict__ hid, const float* __restrict__ encp,
    const float* __restrict__ w_h, const float* __restrict__ w_e,
    const float* __restrict__ rope0, const float* __restrict__ rope1,
    const float* __restrict__ rope2, __bf16* __restrict__ dst, int do_norm_rope) {
  const int R = blockIdx.x, h = blockIdx.y, t = threadIdx.x;
  int stage, p;
  const float* rope;
  if (R < 640)       { stage = 0; p = R;        rope = rope0; }
  else if (R < 1792) { stage = 1; p = R - 640;  rope = rope1; }
  else               { stage = 2; p = R - 1792; rope = rope2; }
  const float* src;
  const float* w;
  if (p < kEnc) {
    src = encp + (size_t)(stage * kEnc + p) * kDim;
    w = w_e;
  } else {
    const int hidoff = (stage == 0) ? 0 : ((stage == 1) ? 512 : 1536);
    src = hid + (size_t)(hidoff + p - kEnc) * kDim;
    w = w_h;
  }
  float x0 = src[h * kHeadDim + 2 * t];
  float x1 = src[h * kHeadDim + 2 * t + 1];
  if (do_norm_rope) {
    float ss = x0 * x0 + x1 * x1;
#pragma unroll
    for (int off = 16; off > 0; off >>= 1) ss += __shfl_xor(ss, off);
    const float rs = rsqrtf(ss * (1.0f / kHeadDim) + 1e-6f);
    x0 = x0 * rs * w[2 * t];
    x1 = x1 * rs * w[2 * t + 1];
    const float* rr = rope + ((size_t)p * 32 + t) * 4;   // [S,1,32,2,2]
    const float y0 = rr[0] * x0 + rr[1] * x1;
    const float y1 = rr[2] * x0 + rr[3] * x1;
    x0 = y0; x1 = y1;
  }
  dst[(size_t)R * kDim + h * kHeadDim + 2 * t]     = (__bf16)x0;
  dst[(size_t)R * kDim + h * kHeadDim + 2 * t + 1] = (__bf16)x1;
}

// ---------------------------------------------------------------------------
// 6) Flash attention: one wave per (stage, head, 16-row q tile).
//    Key blocks of 32: 4 WMMA for QK^T + 4 WMMA for P@V, online softmax f32.
// ---------------------------------------------------------------------------
__global__ __launch_bounds__(32) void attn_kernel(const __bf16* __restrict__ Qs,
                                                  const __bf16* __restrict__ Ks,
                                                  const __bf16* __restrict__ Vs,
                                                  const int* __restrict__ enc_valid,
                                                  float* __restrict__ attn_h,
                                                  float* __restrict__ attn_e) {
  const int tile = blockIdx.x, h = blockIdx.y, lane = threadIdx.x;
  int stage, S, soff, hidoff, tloc;
  if (tile < 40)       { stage = 0; S = 640;  soff = 0;    hidoff = 0;    tloc = tile; }
  else if (tile < 112) { stage = 1; S = 1152; soff = 640;  hidoff = 512;  tloc = tile - 40; }
  else                 { stage = 2; S = 1664; soff = 1792; hidoff = 1536; tloc = tile - 112; }
  const int q0 = tloc * 16;
  const int ev = enc_valid[stage];      // shape (3, B=1)
  const int l15 = lane & 15;
  const int lhi = lane >> 4;

  // Q fragments (16x64 -> two 16x32 A-frags)
  bf16x16 qa[2];
#pragma unroll
  for (int kb = 0; kb < 2; ++kb)
#pragma unroll
    for (int e = 0; e < 16; ++e) {
      const int kk = kb * 32 + ((e < 8) ? 0 : 16) + lhi * 8 + (e & 7);
      qa[kb][e] = Qs[(size_t)(soff + q0 + l15) * kDim + h * kHeadDim + kk];
    }

  f32x8 o[4];
#pragma unroll
  for (int t = 0; t < 4; ++t) o[t] = f32x8{0, 0, 0, 0, 0, 0, 0, 0};
  float mrow[8], lrow[8];
#pragma unroll
  for (int v = 0; v < 8; ++v) { mrow[v] = kNegInf; lrow[v] = 0.f; }

  __shared__ __bf16 Ps[16][40];

  for (int kblk = 0; kblk < S; kblk += 32) {
    f32x8 s0 = f32x8{0, 0, 0, 0, 0, 0, 0, 0};
    f32x8 s1 = f32x8{0, 0, 0, 0, 0, 0, 0, 0};
    // QK^T : B[kdim][n] = K[key n][dim kdim]
#pragma unroll
    for (int kb = 0; kb < 2; ++kb) {
      bf16x16 kf0, kf1;
#pragma unroll
      for (int e = 0; e < 16; ++e) {
        const int kk = kb * 32 + ((e < 8) ? 0 : 16) + lhi * 8 + (e & 7);
        kf0[e] = Ks[(size_t)(soff + kblk + l15) * kDim + h * kHeadDim + kk];
        kf1[e] = Ks[(size_t)(soff + kblk + 16 + l15) * kDim + h * kHeadDim + kk];
      }
      s0 = __builtin_amdgcn_wmma_f32_16x16x32_bf16(false, qa[kb], false, kf0, (short)0, s0, false, false);
      s1 = __builtin_amdgcn_wmma_f32_16x16x32_bf16(false, qa[kb], false, kf1, (short)0, s1, false, false);
    }
    // online softmax (rows m = lhi*8+v, key cols = l15 / 16+l15)
    const int j0 = kblk + l15;
    const int j1 = kblk + 16 + l15;
    const bool ok0 = (j0 >= kEnc) || (j0 < ev);
    const bool ok1 = (j1 >= kEnc) || (j1 < ev);
#pragma unroll
    for (int v = 0; v < 8; ++v) {
      float a = ok0 ? s0[v] * kScale : kNegInf;
      float b = ok1 ? s1[v] * kScale : kNegInf;
      float m = fmaxf(a, b);
#pragma unroll
      for (int off = 8; off > 0; off >>= 1) m = fmaxf(m, __shfl_xor(m, off));
      const float mn = fmaxf(mrow[v], m);
      const float co = __expf(mrow[v] - mn);
      const float p0 = __expf(a - mn);
      const float p1 = __expf(b - mn);
      float rs = p0 + p1;
#pragma unroll
      for (int off = 8; off > 0; off >>= 1) rs += __shfl_xor(rs, off);
      lrow[v] = lrow[v] * co + rs;
      mrow[v] = mn;
#pragma unroll
      for (int t = 0; t < 4; ++t) o[t][v] *= co;
      Ps[lhi * 8 + v][l15]      = (__bf16)p0;
      Ps[lhi * 8 + v][16 + l15] = (__bf16)p1;
    }
    __syncthreads();
    // P (16x32) @ V (32x64)
    bf16x16 pa;
#pragma unroll
    for (int e = 0; e < 16; ++e) {
      const int kk = ((e < 8) ? 0 : 16) + lhi * 8 + (e & 7);
      pa[e] = Ps[l15][kk];
    }
#pragma unroll
    for (int t = 0; t < 4; ++t) {
      bf16x16 vb;
#pragma unroll
      for (int e = 0; e < 16; ++e) {
        const int kdim = ((e < 8) ? 0 : 16) + lhi * 8 + (e & 7);
        vb[e] = Vs[(size_t)(soff + kblk + kdim) * kDim + h * kHeadDim + t * 16 + l15];
      }
      o[t] = __builtin_amdgcn_wmma_f32_16x16x32_bf16(false, pa, false, vb, (short)0, o[t], false, false);
    }
    __syncthreads();
  }

  // write out (split enc / hidden rows), normalized by row sums
#pragma unroll
  for (int t = 0; t < 4; ++t) {
#pragma unroll
    for (int v = 0; v < 8; ++v) {
      const int p = q0 + lhi * 8 + v;
      const int d = t * 16 + l15;
      const float val = o[t][v] / lrow[v];
      if (p < kEnc)
        attn_e[(size_t)(stage * kEnc + p) * kDim + h * kHeadDim + d] = val;
      else
        attn_h[(size_t)(hidoff + p - kEnc) * kDim + h * kHeadDim + d] = val;
    }
  }
}

// ---------------------------------------------------------------------------
// 7) gated residual: out = base + gate[stage]*x
// ---------------------------------------------------------------------------
__global__ __launch_bounds__(256) void resid_gate_kernel(const float* __restrict__ base,
                                                         const float* __restrict__ x,
                                                         const float* __restrict__ mod,
                                                         int gate_sel,
                                                         float* __restrict__ out,
                                                         int is_hidden) {
  const int r = blockIdx.x;
  const int stage = is_hidden ? stage_of_hidden_row(r) : (r >> 7);
  const float* g = mod + (size_t)stage * kSixDim + (size_t)gate_sel * kDim;
  const size_t ro = (size_t)r * kDim;
  for (int c = threadIdx.x; c < kDim; c += 256)
    out[ro + c] = base[ro + c] + g[c] * x[ro + c];
}

// ---------------------------------------------------------------------------
// launch
// ---------------------------------------------------------------------------
extern "C" void kernel_launch(void* const* d_in, const int* in_sizes, int n_in,
                              void* d_out, int out_size, void* d_ws, size_t ws_size,
                              hipStream_t stream) {
  (void)in_sizes; (void)n_in; (void)out_size; (void)ws_size;

  const float* hidden = (const float*)d_in[0];
  const float* encin  = (const float*)d_in[1];
  const float* temb   = (const float*)d_in[2];
  const int*   encv   = (const int*)d_in[3];
  const float* rope0  = (const float*)d_in[4];
  const float* rope1  = (const float*)d_in[5];
  const float* rope2  = (const float*)d_in[6];
  const float* ada_w  = (const float*)d_in[7];
  const float* ada_b  = (const float*)d_in[8];
  const float* adc_w  = (const float*)d_in[9];
  const float* adc_b  = (const float*)d_in[10];
  const float* wq = (const float*)d_in[11]; const float* bq = (const float*)d_in[12];
  const float* wk = (const float*)d_in[13]; const float* bk = (const float*)d_in[14];
  const float* wv = (const float*)d_in[15]; const float* bv = (const float*)d_in[16];
  const float* waq = (const float*)d_in[17]; const float* baq = (const float*)d_in[18];
  const float* wak = (const float*)d_in[19]; const float* bak = (const float*)d_in[20];
  const float* wav = (const float*)d_in[21]; const float* bav = (const float*)d_in[22];
  const float* nq_w  = (const float*)d_in[23];
  const float* nk_w  = (const float*)d_in[24];
  const float* naq_w = (const float*)d_in[25];
  const float* nak_w = (const float*)d_in[26];
  const float* wo  = (const float*)d_in[27]; const float* bo  = (const float*)d_in[28];
  const float* wao = (const float*)d_in[29]; const float* bao = (const float*)d_in[30];
  const float* ff_w1 = (const float*)d_in[31]; const float* ff_b1 = (const float*)d_in[32];
  const float* ff_w2 = (const float*)d_in[33]; const float* ff_b2 = (const float*)d_in[34];
  const float* fc_w1 = (const float*)d_in[35]; const float* fc_b1 = (const float*)d_in[36];
  const float* fc_w2 = (const float*)d_in[37]; const float* fc_b2 = (const float*)d_in[38];

  char* ws = (char*)d_ws;
  float*  silu_t = (float*)(ws + OFF_SILU);
  float*  e6     = (float*)(ws + OFF_E6);
  float*  ec     = (float*)(ws + OFF_EC);
  float*  nh     = (float*)(ws + OFF_NH);
  float*  ne     = (float*)(ws + OFF_NE);
  float*  qb     = (float*)(ws + OFF_Q);
  float*  kb     = (float*)(ws + OFF_K);
  float*  vb     = (float*)(ws + OFF_V);
  float*  eqb    = (float*)(ws + OFF_EQ);
  float*  ekb    = (float*)(ws + OFF_EK);
  float*  evb    = (float*)(ws + OFF_EV);
  __bf16* Qsb    = (__bf16*)(ws + OFF_QS);
  __bf16* Ksb    = (__bf16*)(ws + OFF_KS);
  __bf16* Vsb    = (__bf16*)(ws + OFF_VS);
  float*  a_h    = (float*)(ws + OFF_AH);
  float*  a_e    = (float*)(ws + OFF_AE);
  float*  attnh  = (float*)(ws + OFF_ATTNH);
  float*  attne  = (float*)(ws + OFF_ATTNE);
  float*  h1     = (float*)(ws + OFF_H1);
  float*  mod2   = (float*)(ws + OFF_MOD2);
  float*  fout   = (float*)(ws + OFF_FOUT);
  float*  e1     = (float*)(ws + OFF_E1);
  float*  modc2  = (float*)(ws + OFF_MODC2);
  float*  fcout  = (float*)(ws + OFF_FCOUT);
  __bf16* ffg    = (__bf16*)(ws + OFF_FFG);
  __bf16* ffcg   = (__bf16*)(ws + OFF_FFCG);

  float* out_e = (float*)d_out;                                 // [3,128,1536]
  float* out_h = (float*)d_out + (size_t)kStages * kEnc * kDim; // [3072,1536]

  // 1) AdaLN
  silu_kernel<<<dim3((kStages * kDim + 255) / 256), 256, 0, stream>>>(temb, silu_t, kStages * kDim);
  adaln_kernel<<<dim3((kStages * kSixDim + 255) / 256), 256, 0, stream>>>(silu_t, ada_w, ada_b, e6);
  adaln_kernel<<<dim3((kStages * kSixDim + 255) / 256), 256, 0, stream>>>(silu_t, adc_w, adc_b, ec);

  // 2) norm + modulate (shift=idx0, scale=idx1)
  ln_mod_kernel<<<dim3(kTotal), 256, 0, stream>>>(hidden, e6, nh, 0, 1, 1);
  ln_mod_kernel<<<dim3(kStages * kEnc), 256, 0, stream>>>(encin, ec, ne, 0, 1, 0);

  // 3) QKV projections (bf16 WMMA)
  const dim3 gH(kDim / 128, kTotal / 128);          // (12, 24)
  const dim3 gE(kDim / 128, (kStages * kEnc) / 128);// (12, 3)
  gemm_bf16_wmma<false, false, false><<<gH, 256, 0, stream>>>(nh, wq, bq, qb, kTotal, kDim, kDim);
  gemm_bf16_wmma<false, false, false><<<gH, 256, 0, stream>>>(nh, wk, bk, kb, kTotal, kDim, kDim);
  gemm_bf16_wmma<false, false, false><<<gH, 256, 0, stream>>>(nh, wv, bv, vb, kTotal, kDim, kDim);
  gemm_bf16_wmma<false, false, false><<<gE, 256, 0, stream>>>(ne, waq, baq, eqb, kStages * kEnc, kDim, kDim);
  gemm_bf16_wmma<false, false, false><<<gE, 256, 0, stream>>>(ne, wak, bak, ekb, kStages * kEnc, kDim, kDim);
  gemm_bf16_wmma<false, false, false><<<gE, 256, 0, stream>>>(ne, wav, bav, evb, kStages * kEnc, kDim, kDim);

  // 4) RMSNorm + RoPE + bf16 pack into per-stage [S,24,64]
  const dim3 gP(kRowsAll, kHeads);
  pack_qkv_kernel<<<gP, 32, 0, stream>>>(qb, eqb, nq_w, naq_w, rope0, rope1, rope2, Qsb, 1);
  pack_qkv_kernel<<<gP, 32, 0, stream>>>(kb, ekb, nk_w, nak_w, rope0, rope1, rope2, Ksb, 1);
  pack_qkv_kernel<<<gP, 32, 0, stream>>>(vb, evb, nq_w, naq_w, rope0, rope1, rope2, Vsb, 0);

  // 5) attention
  attn_kernel<<<dim3(kQTilesAll, kHeads), 32, 0, stream>>>(Qsb, Ksb, Vsb, encv, a_h, a_e);

  // 6) output projections
  gemm_bf16_wmma<false, false, false><<<gH, 256, 0, stream>>>(a_h, wo, bo, attnh, kTotal, kDim, kDim);
  gemm_bf16_wmma<false, false, false><<<gE, 256, 0, stream>>>(a_e, wao, bao, attne, kStages * kEnc, kDim, kDim);

  // 7) gated residual (gate_msa = idx2)
  resid_gate_kernel<<<dim3(kTotal), 256, 0, stream>>>(hidden, attnh, e6, 2, h1, 1);
  resid_gate_kernel<<<dim3(kStages * kEnc), 256, 0, stream>>>(encin, attne, ec, 2, e1, 0);

  // 8) MLP norm+mod (shift_mlp = idx3, scale_mlp = idx4)
  ln_mod_kernel<<<dim3(kTotal), 256, 0, stream>>>(h1, e6, mod2, 3, 4, 1);
  ln_mod_kernel<<<dim3(kStages * kEnc), 256, 0, stream>>>(e1, ec, modc2, 3, 4, 0);

  // 9) FFN: GELU(x@W1+b1)@W2+b2  (intermediate kept bf16)
  const dim3 gH1(kInner / 128, kTotal / 128);            // (48, 24)
  const dim3 gE1(kInner / 128, (kStages * kEnc) / 128);  // (48, 3)
  gemm_bf16_wmma<false, true, true><<<gH1, 256, 0, stream>>>(mod2, ff_w1, ff_b1, ffg, kTotal, kInner, kDim);
  gemm_bf16_wmma<false, true, true><<<gE1, 256, 0, stream>>>(modc2, fc_w1, fc_b1, ffcg, kStages * kEnc, kInner, kDim);
  gemm_bf16_wmma<true, false, false><<<gH, 256, 0, stream>>>(ffg, ff_w2, ff_b2, fout, kTotal, kDim, kInner);
  gemm_bf16_wmma<true, false, false><<<gE, 256, 0, stream>>>(ffcg, fc_w2, fc_b2, fcout, kStages * kEnc, kDim, kInner);

  // 10) final gated residual (gate_mlp = idx5) -> d_out (e first, then h)
  resid_gate_kernel<<<dim3(kStages * kEnc), 256, 0, stream>>>(e1, fcout, ec, 5, out_e, 0);
  resid_gate_kernel<<<dim3(kTotal), 256, 0, stream>>>(h1, fout, e6, 5, out_h, 1);
}
